// TernaryLeNet5_91010357002967
// MI455X (gfx1250) — matrix operand order repaired and checked by reading, not compiled, not run on hardware
//
#include <hip/hip_runtime.h>
#include <hip/hip_bf16.h>
#include <math.h>

// ---------------------------------------------------------------------------
// Ternary LeNet-5 forward on gfx1250 (MI455X).
// conv2/conv3/fc1 on v_wmma_f32_16x16x32_f16 (f16 in, f32 accumulate).
// conv2: async global->LDS staging + full im2col in LDS, K-outer loop with
// 7-way N-register-blocking so A-fragments never outlive one iteration
// (eliminates the scratch spills seen with the N-outer structure).
// ---------------------------------------------------------------------------

typedef __attribute__((ext_vector_type(16))) _Float16 v16h;
typedef __attribute__((ext_vector_type(8)))  _Float16 v8h;
typedef __attribute__((ext_vector_type(8)))  float    v8f;

#define N_IMG 1024

// ---------------------------------------------------------------------------
// TWN ternarization with optional row padding:
// delta = 0.7*mean|w|; alpha = mean|w| over survivors; q = alpha*sign*mask.
// qh written as [rows x stride] (cols valid per row); qf written compact.
// ---------------------------------------------------------------------------
__global__ void ternarize_kernel(const float* __restrict__ w, int n,
                                 int cols, int stride,
                                 _Float16* qh, float* qf) {
  __shared__ float red[256];
  __shared__ float s_delta, s_num, s_alpha;
  const int t = threadIdx.x;

  float s = 0.f;
  for (int i = t; i < n; i += 256) s += fabsf(w[i]);
  red[t] = s; __syncthreads();
  for (int k = 128; k > 0; k >>= 1) { if (t < k) red[t] += red[t + k]; __syncthreads(); }
  if (t == 0) s_delta = 0.7f * red[0] / (float)n;
  __syncthreads();
  const float delta = s_delta;

  float sa = 0.f, sc = 0.f;
  for (int i = t; i < n; i += 256) {
    float a = fabsf(w[i]);
    if (a > delta) { sa += a; sc += 1.f; }
  }
  red[t] = sa; __syncthreads();
  for (int k = 128; k > 0; k >>= 1) { if (t < k) red[t] += red[t + k]; __syncthreads(); }
  if (t == 0) s_num = red[0];
  __syncthreads();
  red[t] = sc; __syncthreads();
  for (int k = 128; k > 0; k >>= 1) { if (t < k) red[t] += red[t + k]; __syncthreads(); }
  if (t == 0) s_alpha = s_num / fmaxf(red[0], 1.f);
  __syncthreads();
  const float alpha = s_alpha;

  for (int i = t; i < n; i += 256) {
    float wi = w[i];
    float q  = (fabsf(wi) > delta) ? copysignf(alpha, wi) : 0.f;
    if (qh) {
      int r = i / cols, c = i - r * cols;
      qh[(size_t)r * stride + c] = (_Float16)q;
    }
    if (qf) qf[i] = q;
  }
}

__global__ void zero_f16_kernel(_Float16* __restrict__ p, int n) {
  int i = blockIdx.x * 256 + threadIdx.x;
  if (i < n) p[i] = (_Float16)0.0f;
}

// ---------------------------------------------------------------------------
// conv1 (1->32ch, 5x5) + tanh + 2x2 maxpool fused (direct VALU; C=1, K=25
// is too skinny for WMMA and only ~1.3 GF).  p1: f16 NCHW [1024,32,14,14]
// ---------------------------------------------------------------------------
__global__ void conv1_pool_kernel(const float* __restrict__ x,
                                  const float* __restrict__ w1q,
                                  const float* __restrict__ b1,
                                  _Float16* __restrict__ p1) {
  int id = blockIdx.x * 256 + threadIdx.x;           // < 1024*32*14*14
  int px = id % 14; int t = id / 14;
  int py = t % 14;  t /= 14;
  int oc = t % 32;  int n = t / 32;
  const float* xi = x + (size_t)n * 1024;
  const float* wv = w1q + oc * 25;
  const float  bias = b1[oc];
  float best = -INFINITY;
  #pragma unroll
  for (int dy = 0; dy < 2; ++dy)
  #pragma unroll
  for (int dx = 0; dx < 2; ++dx) {
    int oy = 2 * py + dy, ox = 2 * px + dx;
    float acc = bias;
    #pragma unroll
    for (int ky = 0; ky < 5; ++ky)
      #pragma unroll
      for (int kx = 0; kx < 5; ++kx)
        acc += xi[(oy + ky) * 32 + (ox + kx)] * wv[ky * 5 + kx];
    best = fmaxf(best, tanhf(acc));
  }
  p1[id] = (_Float16)best;
}

// ---------------------------------------------------------------------------
// WMMA fragment layouts (wave32, v_wmma_f32_16x16x32_f16):
//  A 16x32: lane m=lane&15, lh=lane>>4; halves = K in
//           [kb+16lh, kb+16lh+16) contiguous (matches B)
//  B 32x16: lane n=lane&15; halves = K in [kb+16lh, kb+16lh+16) contiguous
//  C  f32 : N = lane&15, M = v + 8*lh (v=0..7)
// ---------------------------------------------------------------------------
__device__ __forceinline__ v16h load_a_frag(const _Float16* __restrict__ row,
                                            int kb, int lh) {
  v8h a0 = *(const v8h*)(row + kb + 16 * lh);
  v8h a1 = *(const v8h*)(row + kb + 16 * lh + 8);
  v16h af;
  #pragma unroll
  for (int i = 0; i < 8; ++i) { af[i] = a0[i]; af[8 + i] = a1[i]; }
  return af;
}

// ---------------------------------------------------------------------------
// conv2 WMMA GEMM, one image per block (128 thr = 4 waves = 4 M-tiles):
//   Out[64 x 100] = W2q[64 x 800] * Col[800 x 100] per image
// Stage 1: async b128 copy of the 12.25KB p1 image into LDS (ASYNCcnt).
// Stage 2: build the FULL 800x112 im2col in LDS once (~175KB; <320KB/WGP).
// Stage 3: K-outer loop: one A-fragment (2 global b128) immediately feeds
//   7 WMMAs (B via ds b128) -> 7x A register reuse, no cross-iteration
//   liveness, nothing for LICM to hoist/spill.  Epilogue: bias + tanh.
// h2: f16 [1024,64,10,10]
// ---------------------------------------------------------------------------
__global__ void conv2_wmma_kernel(const _Float16* __restrict__ A,   // 64x800
                                  const _Float16* __restrict__ P1,  // [1024,6272]
                                  const float*    __restrict__ b2,
                                  _Float16* __restrict__ H2) {
  __shared__ __align__(16) _Float16 simg[6272];        // 32*14*14 image
  __shared__ __align__(32) _Float16 colt[112 * 800];   // im2col [pos][k]
  const int n = blockIdx.x;
  const _Float16* img = P1 + (size_t)n * 6272;

  // Async global->LDS staging: 784 x 16B chunks (ASYNCcnt-tracked).
  {
    unsigned lds_base = (unsigned)(size_t)(&simg[0]);
    for (int c = threadIdx.x; c < 784; c += 128) {
      unsigned ldsa = lds_base + (unsigned)c * 16u;
      unsigned goff = (unsigned)c * 16u;
      asm volatile("global_load_async_to_lds_b128 %0, %1, %2"
                   :: "v"(ldsa), "v"(goff), "s"(img) : "memory");
    }
    asm volatile("s_wait_asynccnt 0" ::: "memory");
  }
  __syncthreads();

  // Build full im2col: colt[p][k] = img[c, oy+ky, ox+kx]
  for (int p = 0; p < 112; ++p) {
    int posc = (p < 100) ? p : 99;
    int oy = posc / 10, ox = posc % 10;
    for (int k = threadIdx.x; k < 800; k += 128) {
      int c  = k / 25, rem = k - 25 * c;
      int ky = rem / 5, kx = rem - 5 * ky;
      colt[p * 800 + k] = simg[(c * 14 + oy + ky) * 14 + (ox + kx)];
    }
  }
  __syncthreads();

  const int lane = threadIdx.x & 31;
  const int wave = threadIdx.x >> 5;        // M-tile 0..3
  const int lh   = lane >> 4, lo = lane & 15;
  const _Float16* arp = A + (size_t)(wave * 16 + lo) * 800;

  v8f acc[7] = {};                          // 7 N-tiles (N=100 pad 112)
  #pragma clang loop unroll(disable)
  for (int kb = 0; kb < 800; kb += 32) {
    v16h af = load_a_frag(arp, kb, lh);     // used 7x, dies this iteration
    #pragma unroll
    for (int nt = 0; nt < 7; ++nt) {
      v16h bf = *(const v16h*)(&colt[(nt * 16 + lo) * 800 + kb + 16 * lh]);
      acc[nt] = __builtin_amdgcn_wmma_f32_16x16x32_f16(false, af, false, bf,
                                                       (short)0, acc[nt],
                                                       false, false);
    }
  }

  #pragma unroll
  for (int nt = 0; nt < 7; ++nt) {
    const int pos = nt * 16 + lo;
    if (pos < 100) {
      #pragma unroll
      for (int v = 0; v < 8; ++v) {
        int oc = wave * 16 + v + 8 * lh;    // < 64
        float y = tanhf(acc[nt][v] + b2[oc]);
        H2[((size_t)n * 64 + oc) * 100 + pos] = (_Float16)y;
      }
    }
  }
}

// ---------------------------------------------------------------------------
// 2x2 maxpool h2 [1024,64,10,10] -> p2 [1024,64,5,5] (f16)
// ---------------------------------------------------------------------------
__global__ void maxpool2_kernel(const _Float16* __restrict__ H2,
                                _Float16* __restrict__ P2) {
  int id = blockIdx.x * 256 + threadIdx.x;           // < 1024*64*25
  int px = id % 5; int t = id / 5;
  int py = t % 5;  t /= 5;
  int oc = t % 64; int n = t / 64;
  const _Float16* b = H2 + ((size_t)n * 64 + oc) * 100;
  float m = fmaxf(fmaxf((float)b[(2 * py) * 10 + 2 * px],
                        (float)b[(2 * py) * 10 + 2 * px + 1]),
                  fmaxf((float)b[(2 * py + 1) * 10 + 2 * px],
                        (float)b[(2 * py + 1) * 10 + 2 * px + 1]));
  P2[id] = (_Float16)m;
}

// ---------------------------------------------------------------------------
// conv3: 5x5 on 5x5 input -> im2col is the flat NCHW vector.
//   Out[128pad x 1024] = W3q[128 x 1600 zero-padded] * P2flat[1600 x 1024]
// Guard-free K loop; all fragment loads aligned b128, used once each.
// h3: [1024,128pad]
// ---------------------------------------------------------------------------
__global__ void conv3_wmma_kernel(const _Float16* __restrict__ A,   // 128x1600
                                  const _Float16* __restrict__ P2,  // [1024,1600]
                                  const float*    __restrict__ b3,
                                  _Float16* __restrict__ H3) {     // [1024,128]
  const int lane = threadIdx.x & 31;
  const int wave = threadIdx.x >> 5;        // M-tile 0..7
  const int lh   = lane >> 4, lo = lane & 15;
  const int col  = blockIdx.x * 16 + lo;    // image index
  const _Float16* cp  = P2 + (size_t)col * 1600;
  const _Float16* arp = A + (size_t)(wave * 16 + lo) * 1600;

  v8f acc = {};
  #pragma clang loop unroll_count(4)
  for (int kb = 0; kb < 1600; kb += 32) {
    __builtin_prefetch(cp + kb + 128, 0, 0);
    v16h af = load_a_frag(arp, kb, lh);
    v16h bf = *(const v16h*)(cp + kb + 16 * lh);   // 32B aligned contiguous
    acc = __builtin_amdgcn_wmma_f32_16x16x32_f16(false, af, false, bf,
                                                 (short)0, acc, false, false);
  }
  #pragma unroll
  for (int v = 0; v < 8; ++v) {
    int m = wave * 16 + v + 8 * lh;
    if (m < 120) {
      float y = tanhf(acc[v] + b3[m]);
      H3[(size_t)col * 128 + m] = (_Float16)y;   // pad cols stay zero
    }
  }
}

// ---------------------------------------------------------------------------
// fc1: Out[96pad x 1024] = FW1q[96 x 128 zero-padded] * H3^T[128 x 1024]
// Guard-free K loop (pads are zero; 4 steps, full unroll is cheap).
// h4: f16 [1024,96pad] (pads zeroed)
// ---------------------------------------------------------------------------
__global__ void fc1_wmma_kernel(const _Float16* __restrict__ A,   // 96x128
                                const _Float16* __restrict__ H3,  // [1024,128]
                                const float*    __restrict__ fb1,
                                _Float16* __restrict__ H4) {      // [1024,96]
  const int lane = threadIdx.x & 31;
  const int wave = threadIdx.x >> 5;        // M-tile 0..5
  const int lh   = lane >> 4, lo = lane & 15;
  const int col  = blockIdx.x * 16 + lo;    // image index
  const _Float16* cp  = H3 + (size_t)col * 128;
  const _Float16* arp = A + (size_t)(wave * 16 + lo) * 128;

  v8f acc = {};
  #pragma unroll
  for (int kb = 0; kb < 128; kb += 32) {
    v16h af = load_a_frag(arp, kb, lh);
    v16h bf = *(const v16h*)(cp + kb + 16 * lh);
    acc = __builtin_amdgcn_wmma_f32_16x16x32_f16(false, af, false, bf,
                                                 (short)0, acc, false, false);
  }
  #pragma unroll
  for (int v = 0; v < 8; ++v) {
    int m = wave * 16 + v + 8 * lh;
    if (m < 84) {
      float y = tanhf(acc[v] + fb1[m]);
      H4[(size_t)col * 96 + m] = (_Float16)y;   // pad cols stay zero
    }
  }
}

// ---------------------------------------------------------------------------
// fc2: logits = H4[1024,96pad] * FW2q^T[10,96pad] + fb2  (wide v8h loads)
// ---------------------------------------------------------------------------
__global__ void fc2_kernel(const _Float16* __restrict__ H4,   // [1024,96]
                           const _Float16* __restrict__ W,    // [10,96]
                           const float*    __restrict__ fb2,
                           float* __restrict__ logits) {
  int id = blockIdx.x * 256 + threadIdx.x;
  if (id >= N_IMG * 10) return;
  int cls = id % 10, n = id / 10;
  const v8h* hv = (const v8h*)(H4 + (size_t)n * 96);    // 192B rows: aligned
  const v8h* wv = (const v8h*)(W  + (size_t)cls * 96);
  float s = fb2[cls];
  #pragma unroll
  for (int j = 0; j < 12; ++j) {
    v8h a = hv[j], b = wv[j];
    #pragma unroll
    for (int e = 0; e < 8; ++e) s += (float)a[e] * (float)b[e];
  }
  logits[id] = s;
}

__global__ void softmax_kernel(const float* __restrict__ logits,
                               float* __restrict__ probs) {
  int n = blockIdx.x * 256 + threadIdx.x;
  if (n >= N_IMG) return;
  const float* l = logits + n * 10;
  float m = -INFINITY;
  #pragma unroll
  for (int c = 0; c < 10; ++c) m = fmaxf(m, l[c]);
  float e[10], s = 0.f;
  #pragma unroll
  for (int c = 0; c < 10; ++c) { e[c] = expf(l[c] - m); s += e[c]; }
  float inv = 1.f / s;
  #pragma unroll
  for (int c = 0; c < 10; ++c) probs[n * 10 + c] = e[c] * inv;
}

// ---------------------------------------------------------------------------
extern "C" void kernel_launch(void* const* d_in, const int* in_sizes, int n_in,
                              void* d_out, int out_size, void* d_ws, size_t ws_size,
                              hipStream_t stream) {
  const float* x   = (const float*)d_in[0];
  const float* w1  = (const float*)d_in[1];
  const float* b1  = (const float*)d_in[2];
  const float* w2  = (const float*)d_in[3];
  const float* b2  = (const float*)d_in[4];
  const float* w3  = (const float*)d_in[5];
  const float* b3  = (const float*)d_in[6];
  const float* fw1 = (const float*)d_in[7];
  const float* fb1 = (const float*)d_in[8];
  const float* fw2 = (const float*)d_in[9];
  const float* fb2 = (const float*)d_in[10];

  // Workspace carve-out (running cursor, 256B aligned); total ~31 MB.
  char* ws = (char*)d_ws;
  size_t off = 0;
  auto carve = [&](size_t bytes) -> char* {
    char* p = ws + off;
    off = (off + bytes + 255) & ~(size_t)255;
    return p;
  };
  float*    w1q  = (float*)   carve(800          * sizeof(float));
  _Float16* w2q  = (_Float16*)carve(64  * 800    * sizeof(_Float16));
  _Float16* w3q  = (_Float16*)carve(128 * 1600   * sizeof(_Float16));  // M pad
  _Float16* fw1q = (_Float16*)carve(96  * 128    * sizeof(_Float16));  // M,K pad
  _Float16* fw2q = (_Float16*)carve(10  * 96     * sizeof(_Float16));  // K pad
  _Float16* p1   = (_Float16*)carve((size_t)N_IMG * 6272 * sizeof(_Float16));
  _Float16* h2   = (_Float16*)carve((size_t)N_IMG * 6400 * sizeof(_Float16));
  _Float16* p2   = (_Float16*)carve((size_t)N_IMG * 1600 * sizeof(_Float16));
  _Float16* h3   = (_Float16*)carve((size_t)N_IMG * 128  * sizeof(_Float16)); // K pad
  _Float16* h4   = (_Float16*)carve((size_t)N_IMG * 96   * sizeof(_Float16)); // K pad

  float* logits = (float*)d_out;
  float* probs  = (float*)d_out + N_IMG * 10;

  // 0. zero the padded regions (deterministic each call)
  zero_f16_kernel<<<(12800 + 255) / 256, 256, 0, stream>>>(w3q + 192000, 12800);
  zero_f16_kernel<<<(96 * 128 + 255) / 256, 256, 0, stream>>>(fw1q, 96 * 128);
  zero_f16_kernel<<<(10 * 96 + 255) / 256, 256, 0, stream>>>(fw2q, 10 * 96);
  zero_f16_kernel<<<(N_IMG * 128) / 256, 256, 0, stream>>>(h3, N_IMG * 128);
  zero_f16_kernel<<<(N_IMG * 96) / 256, 256, 0, stream>>>(h4, N_IMG * 96);

  // 1. ternarize all weight tensors
  ternarize_kernel<<<1, 256, 0, stream>>>(w1,  800,    800,  800,  nullptr, w1q);
  ternarize_kernel<<<1, 256, 0, stream>>>(w2,  51200,  800,  800,  w2q,  nullptr);
  ternarize_kernel<<<1, 256, 0, stream>>>(w3,  192000, 1600, 1600, w3q,  nullptr);
  ternarize_kernel<<<1, 256, 0, stream>>>(fw1, 10080,  120,  128,  fw1q, nullptr);
  ternarize_kernel<<<1, 256, 0, stream>>>(fw2, 840,    84,   96,   fw2q, nullptr);

  // 2. conv1 + tanh + pool (direct)
  conv1_pool_kernel<<<(N_IMG * 32 * 14 * 14) / 256, 256, 0, stream>>>(x, w1q, b1, p1);

  // 3. conv2 via WMMA + full LDS im2col (64 x 800 x 102400)
  conv2_wmma_kernel<<<N_IMG, 128, 0, stream>>>(w2q, p1, b2, h2);

  // 4. pool, conv3 via WMMA GEMM (120 x 1600 x 1024)
  maxpool2_kernel<<<(N_IMG * 64 * 25) / 256, 256, 0, stream>>>(h2, p2);
  conv3_wmma_kernel<<<N_IMG / 16, 256, 0, stream>>>(w3q, p2, b3, h3);

  // 5. fc1 via WMMA GEMM (84 x 120 x 1024)
  fc1_wmma_kernel<<<N_IMG / 16, 192, 0, stream>>>(fw1q, h3, fb1, h4);

  // 6. fc2 + softmax
  fc2_kernel<<<(N_IMG * 10 + 255) / 256, 256, 0, stream>>>(h4, fw2q, fb2, logits);
  softmax_kernel<<<(N_IMG + 255) / 256, 256, 0, stream>>>(logits, probs);
}
